// ssm_LTCCell_90108413870904
// MI455X (gfx1250) — matrix operand order repaired and checked by reading, not compiled, not run on hardware
//
#include <hip/hip_runtime.h>
#include <hip/hip_bf16.h>
#include <math.h>

#define Bsz 64
#define Tsz 1024
#define Fsz 512
#define Isz 256
#define Ssz 128
#define Msz 64
#define ODE_UNFOLDS 6
#define EPS_ 1e-8f

typedef __attribute__((ext_vector_type(16))) _Float16 v16h;
typedef __attribute__((ext_vector_type(8)))  float    v8f;

// gfx1250 async global->LDS b128 builtin takes (v4i AS1*, v4i AS3*, imm, imm)
typedef int v4i_ __attribute__((__vector_size__(16)));
typedef __attribute__((address_space(1))) v4i_ ga_b128_t;   // global
typedef __attribute__((address_space(3))) v4i_ ls_b128_t;   // LDS

#if __has_builtin(__builtin_amdgcn_global_load_async_to_lds_b128) && \
    __has_builtin(__builtin_amdgcn_s_wait_asynccnt)
#define HAVE_ASYNC_LDS 1
#else
#define HAVE_ASYNC_LDS 0
#endif

// Branch-free tanh: 1 - 2/(1+exp(2x)); saturates to +/-1, single v_exp_f32.
__device__ __forceinline__ float fast_tanh(float x) {
    return 1.0f - 2.0f / (1.0f + __expf(2.0f * x));
}
__device__ __forceinline__ float fast_sigmoid(float x) {
    return 1.0f / (1.0f + __expf(-x));
}

// ---------------------------------------------------------------------------
// Pack Bw (K x N f32, row-major) into WMMA B-fragment order, f16:
//   Bpack[((chunk*N + col) * 32) + half]   (chunk = kc/32)
//   halves 0..15  = K kc..kc+15   (consumed by lanes 0-15)
//   halves 16..31 = K kc+16..kc+31 (consumed by lanes 16-31)
// One thread per (chunk, col, half-group of 16).
// ---------------------------------------------------------------------------
__global__ void pack_b_kernel(const float* __restrict__ Bw,
                              _Float16* __restrict__ Bp, int K, int N)
{
    const int idx   = blockIdx.x * blockDim.x + threadIdx.x;
    const int total = (K / 32) * N * 2;
    if (idx >= total) return;
    const int halfgrp = idx & 1;
    const int col     = (idx >> 1) % N;
    const int chunk   = (idx >> 1) / N;
    const int kc      = chunk * 32 + halfgrp * 16;
    _Float16* dst = Bp + ((size_t)(chunk * N + col) * 32) + halfgrp * 16;
#pragma unroll
    for (int h = 0; h < 16; ++h)
        dst[h] = (_Float16)Bw[(size_t)(kc + h) * N + col];
}

// ---------------------------------------------------------------------------
// WMMA GEMM: out = [tanh](A @ B + bias). Each wave32 computes a 16x64 strip:
// one A fragment feeds 4 accumulators (4x v_wmma per K-step).
// A: Mrows x K f32 row-major (vectorized float4 loads + inline cvt to f16).
// Bpack: pre-packed f16 fragments (contiguous 32B per fragment per lane).
// Columns < splitN -> out0 (ld=splitN), columns >= splitN -> out1 (ld=N-splitN)
// ---------------------------------------------------------------------------
__global__ void gemm_wmma_kernel(const float* __restrict__ A,
                                 const _Float16* __restrict__ Bpack,
                                 const float* __restrict__ bias,
                                 float* __restrict__ out0,
                                 float* __restrict__ out1,
                                 int Mrows, int K, int N, int splitN, int doTanh)
{
    const int lane    = threadIdx.x & 31;
    const int wave    = blockIdx.x * (blockDim.x >> 5) + (threadIdx.x >> 5);
    const int nstrips = N >> 6;                       // 64-wide output strips
    const int tile_m  = wave / nstrips;
    const int strip   = wave % nstrips;
    if (tile_m * 16 >= Mrows) return;

    const bool hi   = lane >= 16;
    const int  row  = tile_m * 16 + (lane & 15);
    const int  col0 = strip * 64 + (lane & 15);

    v8f acc0 = {}, acc1 = {}, acc2 = {}, acc3 = {};

    for (int kc = 0; kc < K; kc += 32) {
        const int chunk = kc >> 5;
        // A-matrix 16-bit 16x32 layout (contiguous k, matches packed B):
        // lane<16:  halves0-15 = K kc..kc+15
        // lane>=16: halves0-15 = K kc+16..kc+31
        const int akb = kc + (hi ? 16 : 0);
        const float* Ap = A + (size_t)row * K + akb;      // 32B aligned
        const float4 x0 = *(const float4*)(Ap);
        const float4 x1 = *(const float4*)(Ap + 4);
        const float4 y0 = *(const float4*)(Ap + 8);
        const float4 y1 = *(const float4*)(Ap + 12);
        v16h a;
        a[0]  = (_Float16)x0.x; a[1]  = (_Float16)x0.y;
        a[2]  = (_Float16)x0.z; a[3]  = (_Float16)x0.w;
        a[4]  = (_Float16)x1.x; a[5]  = (_Float16)x1.y;
        a[6]  = (_Float16)x1.z; a[7]  = (_Float16)x1.w;
        a[8]  = (_Float16)y0.x; a[9]  = (_Float16)y0.y;
        a[10] = (_Float16)y0.z; a[11] = (_Float16)y0.w;
        a[12] = (_Float16)y1.x; a[13] = (_Float16)y1.y;
        a[14] = (_Float16)y1.z; a[15] = (_Float16)y1.w;

        // B fragments: each is 32 contiguous bytes in packed layout.
        // Col tiles are 16 cols apart = 16*32 halves = 32 v16h units apart.
        const v16h* bp = (const v16h*)(Bpack + ((size_t)(chunk * N + col0) * 32)
                                             + (hi ? 16 : 0));
        const v16h b0 = bp[0];
        const v16h b1 = bp[32];
        const v16h b2 = bp[64];
        const v16h b3 = bp[96];

        acc0 = __builtin_amdgcn_wmma_f32_16x16x32_f16(false, a, false, b0,
                                                      (short)0, acc0, false, false);
        acc1 = __builtin_amdgcn_wmma_f32_16x16x32_f16(false, a, false, b1,
                                                      (short)0, acc1, false, false);
        acc2 = __builtin_amdgcn_wmma_f32_16x16x32_f16(false, a, false, b2,
                                                      (short)0, acc2, false, false);
        acc3 = __builtin_amdgcn_wmma_f32_16x16x32_f16(false, a, false, b3,
                                                      (short)0, acc3, false, false);
    }

    // D layout: VGPR r holds row (tile_m*16 + r + (hi?8:0)), col fixed per lane.
    const int rbase = tile_m * 16 + (hi ? 8 : 0);
#pragma unroll
    for (int nt = 0; nt < 4; ++nt) {
        const int col = col0 + nt * 16;
        const v8f acc = (nt == 0) ? acc0 : (nt == 1) ? acc1 : (nt == 2) ? acc2 : acc3;
        const float bv = bias[col];
#pragma unroll
        for (int r = 0; r < 8; ++r) {
            float val = acc[r] + bv;
            if (doTanh) val = fast_tanh(val);
            const int m = rbase + r;
            if (col < splitN)
                out0[(size_t)m * splitN + col] = val;
            else
                out1[(size_t)m * (N - splitN) + (col - splitN)] = val;
        }
    }
}

// ---------------------------------------------------------------------------
// Softmax over the TIME axis of gate (B,T,S): one block per (b,s) pair.
// ---------------------------------------------------------------------------
__global__ void softmax_time_kernel(float* __restrict__ gate)
{
    const int bs = blockIdx.x;            // 0 .. B*S-1
    const int b  = bs / Ssz;
    const int s  = bs % Ssz;
    float* g = gate + (size_t)b * Tsz * Ssz + s;

    __shared__ float red[256];
    const int tid = threadIdx.x;

    float mx = -INFINITY;
    for (int t = tid; t < Tsz; t += blockDim.x)
        mx = fmaxf(mx, g[(size_t)t * Ssz]);
    red[tid] = mx;
    __syncthreads();
    for (int o = blockDim.x >> 1; o > 0; o >>= 1) {
        if (tid < o) red[tid] = fmaxf(red[tid], red[tid + o]);
        __syncthreads();
    }
    mx = red[0];
    __syncthreads();

    float sum = 0.f;
    for (int t = tid; t < Tsz; t += blockDim.x)
        sum += __expf(g[(size_t)t * Ssz] - mx);
    red[tid] = sum;
    __syncthreads();
    for (int o = blockDim.x >> 1; o > 0; o >>= 1) {
        if (tid < o) red[tid] += red[tid + o];
        __syncthreads();
    }
    const float inv = 1.0f / red[0];

    for (int t = tid; t < Tsz; t += blockDim.x)
        g[(size_t)t * Ssz] = __expf(g[(size_t)t * Ssz] - mx) * inv;
}

// ---------------------------------------------------------------------------
// Sensory synapse pre-computation (hoisted out of scan): one block per (b,t),
// 128 threads = one output state j each. hidden row broadcast via LDS using
// gfx1250 async global->LDS copy (ASYNCcnt).
// ---------------------------------------------------------------------------
__global__ void sensory_kernel(const float* __restrict__ hidden,
                               const float* __restrict__ sw,
                               const float* __restrict__ ssig,
                               const float* __restrict__ smu,
                               const float* __restrict__ serev,
                               const float* __restrict__ smask,
                               float* __restrict__ wnum,
                               float* __restrict__ wden)
{
    const int bt = blockIdx.x;            // 0 .. B*T-1
    const int j  = threadIdx.x;           // 0 .. S-1
    __shared__ float h[Isz];
    const float* hp = hidden + (size_t)bt * Isz;

#if HAVE_ASYNC_LDS
    if (j < Isz / 4) {                    // 64 lanes x 16B = 1KB row
        __builtin_amdgcn_global_load_async_to_lds_b128(
            (ga_b128_t*)(hp + j * 4), (ls_b128_t*)(h + j * 4), 0, 0);
    }
    __builtin_amdgcn_s_wait_asynccnt(0);
    __syncthreads();
#else
    for (int i = j; i < Isz; i += Ssz) h[i] = hp[i];
    __syncthreads();
#endif

    float num = 0.f, den = 0.f;
#pragma unroll 4
    for (int i = 0; i < Isz; ++i) {
        const int idx = i * Ssz + j;
        const float x  = (h[i] - smu[idx]) * ssig[idx];
        const float sg = sw[idx] * smask[idx] * fast_sigmoid(x);
        num += sg * serev[idx];
        den += sg;
    }
    wnum[(size_t)bt * Ssz + j] = num;
    wden[(size_t)bt * Ssz + j] = den;
}

// ---------------------------------------------------------------------------
// Sequential ODE scan: one persistent block per batch (batches independent).
// All four SxS parameter tensors (256 KB) live in CDNA5's 320 KB LDS; mu and
// sigma are staged with async global->LDS copies (ASYNCcnt), w*mask and
// w*mask*erev need ALU so they go through VGPRs.
// ---------------------------------------------------------------------------
__global__ void scan_kernel(const float* __restrict__ w,
                            const float* __restrict__ sigma,
                            const float* __restrict__ mu,
                            const float* __restrict__ erev,
                            const float* __restrict__ mask,
                            const float* __restrict__ gleak,
                            const float* __restrict__ vleak,
                            const float* __restrict__ cm,
                            const float* __restrict__ ode0,
                            const float* __restrict__ wnum_s,
                            const float* __restrict__ wden_s,
                            const float* __restrict__ Cbuf,
                            const float* __restrict__ gate,
                            float* __restrict__ fused_out,
                            float* __restrict__ final_state)
{
    extern __shared__ float smem[];
    float* wm  = smem;                     // w * mask          (S*S)
    float* we  = wm  + Ssz * Ssz;          // w * mask * erev   (S*S)
    float* mus = we  + Ssz * Ssz;          // mu                (S*S)
    float* sgs = mus + Ssz * Ssz;          // sigma             (S*S)
    float* vsh = sgs + Ssz * Ssz;          // state broadcast   (S)

    const int b = blockIdx.x;
    const int j = threadIdx.x;             // 0 .. S-1

#if HAVE_ASYNC_LDS
    // mu/sigma: pure copies -> async DMA into LDS (ASYNCcnt).
    for (int e4 = j; e4 < (Ssz * Ssz) / 4; e4 += Ssz) {
        __builtin_amdgcn_global_load_async_to_lds_b128(
            (ga_b128_t*)(mu + e4 * 4), (ls_b128_t*)(mus + e4 * 4), 0, 0);
        __builtin_amdgcn_global_load_async_to_lds_b128(
            (ga_b128_t*)(sigma + e4 * 4), (ls_b128_t*)(sgs + e4 * 4), 0, 0);
    }
    for (int e = j; e < Ssz * Ssz; e += Ssz) {
        const float wv = w[e] * mask[e];
        wm[e] = wv;
        we[e] = wv * erev[e];
    }
    __builtin_amdgcn_s_wait_asynccnt(0);
#else
    for (int e = j; e < Ssz * Ssz; e += Ssz) {
        const float wv = w[e] * mask[e];
        wm[e]  = wv;
        we[e]  = wv * erev[e];
        mus[e] = mu[e];
        sgs[e] = sigma[e];
    }
#endif

    float v = ode0[b * Ssz + j];
    const float cmt = cm[j] * (float)ODE_UNFOLDS;   // cm / (ELAPSED/UNFOLDS)
    const float gl  = gleak[j];
    const float glv = gl * vleak[j];

    vsh[j] = v;
    __syncthreads();

    for (int t = 0; t < Tsz; ++t) {
        const size_t base = ((size_t)b * Tsz + t) * Ssz + j;
        const float wns = wnum_s[base];
        const float wds = wden_s[base];
        const float Ct  = Cbuf[base];
        const float gt  = gate[base];
        if (t + 1 < Tsz) {                  // global_prefetch_b8 for next step
            __builtin_prefetch(&wnum_s[base + Ssz], 0, 1);
            __builtin_prefetch(&wden_s[base + Ssz], 0, 1);
            __builtin_prefetch(&Cbuf[base + Ssz],   0, 1);
            __builtin_prefetch(&gate[base + Ssz],   0, 1);
        }

        for (int u = 0; u < ODE_UNFOLDS; ++u) {
            float num = wns, den = wds;
#pragma unroll 4
            for (int i = 0; i < Ssz; ++i) {
                const int idx = i * Ssz + j;
                const float x  = (vsh[i] - mus[idx]) * sgs[idx];
                const float sg = fast_sigmoid(x);
                num += we[idx] * sg;
                den += wm[idx] * sg;
            }
            num += cmt * v + glv;
            den += cmt + gl;
            v = num / (den + EPS_);
            __syncthreads();
            vsh[j] = v;
            __syncthreads();
        }

        fused_out[base] = v * Ct * fast_sigmoid(gt);
    }

    final_state[b * Ssz + j] = v;
}

// ---------------------------------------------------------------------------
extern "C" void kernel_launch(void* const* d_in, const int* in_sizes, int n_in,
                              void* d_out, int out_size, void* d_ws, size_t ws_size,
                              hipStream_t stream)
{
    (void)in_sizes; (void)n_in; (void)out_size; (void)ws_size;

    const float* inputs    = (const float*)d_in[0];   // (B,T,F)
    const float* ode_state = (const float*)d_in[1];   // (B,S)
    const float* W_in      = (const float*)d_in[2];   // (F, I+S)
    const float* b_in      = (const float*)d_in[3];   // (I+S)
    const float* W_c       = (const float*)d_in[4];   // (I, S)
    const float* b_c       = (const float*)d_in[5];   // (S)
    const float* W_out     = (const float*)d_in[6];   // (S, M)
    const float* b_out     = (const float*)d_in[7];   // (M)
    const float* gleak     = (const float*)d_in[8];
    const float* vleak     = (const float*)d_in[9];
    const float* cm        = (const float*)d_in[10];
    const float* w         = (const float*)d_in[11];
    const float* sigma     = (const float*)d_in[12];
    const float* mu        = (const float*)d_in[13];
    const float* sw        = (const float*)d_in[14];
    const float* ssig      = (const float*)d_in[15];
    const float* smu       = (const float*)d_in[16];
    const float* erev      = (const float*)d_in[17];
    const float* serev     = (const float*)d_in[18];
    const float* mask      = (const float*)d_in[19];
    const float* smask     = (const float*)d_in[20];

    // Output layout: outputs (B,T,M) | final_state (B,S) | fused_out (B,T,S)
    float* outputs     = (float*)d_out;
    float* final_state = outputs + (size_t)Bsz * Tsz * Msz;
    float* fused_out   = final_state + (size_t)Bsz * Ssz;

    // Workspace layout (f32 + packed f16 weight fragments)
    float* hidden  = (float*)d_ws;                              // B*T*I
    float* gatebuf = hidden + (size_t)Bsz * Tsz * Isz;          // B*T*S
    float* Cbuf    = gatebuf + (size_t)Bsz * Tsz * Ssz;         // B*T*S
    float* wnum_s  = Cbuf + (size_t)Bsz * Tsz * Ssz;            // B*T*S
    float* wden_s  = wnum_s + (size_t)Bsz * Tsz * Ssz;          // B*T*S
    _Float16* Bp1  = (_Float16*)(wden_s + (size_t)Bsz * Tsz * Ssz);
    _Float16* Bp2  = Bp1 + (size_t)(Fsz / 32) * (Isz + Ssz) * 32;
    _Float16* Bp3  = Bp2 + (size_t)(Isz / 32) * Ssz * 32;

    const int Mrows = Bsz * Tsz;                                // 65536

    // 0) pack weights into WMMA B-fragment layout (f16)
    {
        const int t1 = (Fsz / 32) * (Isz + Ssz) * 2;
        pack_b_kernel<<<(t1 + 255) / 256, 256, 0, stream>>>(W_in, Bp1, Fsz, Isz + Ssz);
        const int t2 = (Isz / 32) * Ssz * 2;
        pack_b_kernel<<<(t2 + 255) / 256, 256, 0, stream>>>(W_c, Bp2, Isz, Ssz);
        const int t3 = (Ssz / 32) * Msz * 2;
        pack_b_kernel<<<(t3 + 255) / 256, 256, 0, stream>>>(W_out, Bp3, Ssz, Msz);
    }

    // 1) proj = tanh(X @ W_in + b_in), split into hidden / gate
    {
        const int N = Isz + Ssz;                                // 384
        const int waves  = (Mrows / 16) * (N / 64);
        const int blocks = waves / 8;
        gemm_wmma_kernel<<<blocks, 256, 0, stream>>>(
            inputs, Bp1, b_in, hidden, gatebuf, Mrows, Fsz, N, Isz, 1);
    }

    // 2) softmax over time axis of gate
    softmax_time_kernel<<<Bsz * Ssz, 256, 0, stream>>>(gatebuf);

    // 3) C = hidden @ W_c + b_c
    {
        const int N = Ssz;                                      // 128
        const int waves  = (Mrows / 16) * (N / 64);
        const int blocks = waves / 8;
        gemm_wmma_kernel<<<blocks, 256, 0, stream>>>(
            hidden, Bp2, b_c, Cbuf, Cbuf, Mrows, Isz, N, N, 0);
    }

    // 4) sensory synapse sums for every (b,t) — hoisted out of the scan
    sensory_kernel<<<Bsz * Tsz, Ssz, 0, stream>>>(
        hidden, sw, ssig, smu, serev, smask, wnum_s, wden_s);

    // 5) sequential ODE scan; 256 KB of S×S params + state in LDS
    {
        const size_t shbytes = (size_t)(4 * Ssz * Ssz + Ssz) * sizeof(float);
        scan_kernel<<<Bsz, Ssz, shbytes, stream>>>(
            w, sigma, mu, erev, mask, gleak, vleak, cm, ode_state,
            wnum_s, wden_s, Cbuf, gatebuf, fused_out, final_state);
    }

    // 6) outputs = fused @ W_out + b_out
    {
        const int N = Msz;                                      // 64
        const int waves  = (Mrows / 16) * (N / 64);
        const int blocks = waves / 8;
        gemm_wmma_kernel<<<blocks, 256, 0, stream>>>(
            fused_out, Bp3, b_out, outputs, outputs, Mrows, Ssz, N, N, 0);
    }
}
